// MetaSelector_37125697306649
// MI455X (gfx1250) — matrix operand order repaired and checked by reading, not compiled, not run on hardware
//
#include <hip/hip_runtime.h>

typedef __bf16 v16bf __attribute__((ext_vector_type(16)));
typedef float  v8f   __attribute__((ext_vector_type(8)));
typedef unsigned int v8u __attribute__((ext_vector_type(8)));

#define B_N   4096
#define F_DIM 3072
#define NC    1000
#define E_NUM 8

__device__ __forceinline__ unsigned short f2bf(float f) {
    unsigned int u = __builtin_bit_cast(unsigned int, f);
    unsigned int r = u + 0x7FFFu + ((u >> 16) & 1u);
    return (unsigned short)(r >> 16);
}

// Load 16 bf16 elements as two 16-byte chunks: p[0..7] and p[gap..gap+7].
__device__ __forceinline__ v16bf load_frag(const unsigned short* p, int gap) {
    uint4 lo = *(const uint4*)(p);
    uint4 hi = *(const uint4*)(p + gap);
    v8u u;
    u[0] = lo.x; u[1] = lo.y; u[2] = lo.z; u[3] = lo.w;
    u[4] = hi.x; u[5] = hi.y; u[6] = hi.z; u[7] = hi.w;
    return __builtin_bit_cast(v16bf, u);
}

// ---------------- kernel 1: x fp32 -> bf16 ----------------
__global__ void convert_x_kernel(const float* __restrict__ x,
                                 unsigned short* __restrict__ xbf) {
    size_t i = ((size_t)blockIdx.x * blockDim.x + threadIdx.x) * 8;
    float4 a = *(const float4*)(x + i);
    float4 b = *(const float4*)(x + i + 4);
    uint4 u;
    u.x = (unsigned)f2bf(a.x) | ((unsigned)f2bf(a.y) << 16);
    u.y = (unsigned)f2bf(a.z) | ((unsigned)f2bf(a.w) << 16);
    u.z = (unsigned)f2bf(b.x) | ((unsigned)f2bf(b.y) << 16);
    u.w = (unsigned)f2bf(b.z) | ((unsigned)f2bf(b.w) << 16);
    *(uint4*)(xbf + i) = u;
}

// ------- kernel 2: ens_w [E,F,NC] fp32 -> wt [E,NC,F] bf16 (tiled transpose) -------
__global__ void convert_transpose_w_kernel(const float* __restrict__ w,
                                           unsigned short* __restrict__ wt) {
    __shared__ float tile[32][33];
    int e = blockIdx.z;
    int f0 = blockIdx.x * 32;
    int c0 = blockIdx.y * 32;
    int tx = threadIdx.x, ty = threadIdx.y;
#pragma unroll
    for (int j = 0; j < 4; ++j) {
        int f = f0 + ty + j * 8;
        int c = c0 + tx;
        float v = 0.f;
        if (c < NC) v = w[((size_t)e * F_DIM + f) * NC + c];
        tile[ty + j * 8][tx] = v;
    }
    __syncthreads();
#pragma unroll
    for (int j = 0; j < 4; ++j) {
        int c = c0 + ty + j * 8;
        int f = f0 + tx;
        if (c < NC)
            wt[((size_t)e * NC + c) * F_DIM + f] = f2bf(tile[tx][ty + j * 8]);
    }
}

// ---------------- kernel 3: selector CNN + top-k routing ----------------
__global__ __launch_bounds__(256) void selector_kernel(
    const float* __restrict__ x,
    const float* __restrict__ c1w, const float* __restrict__ c1b,
    const float* __restrict__ c2w, const float* __restrict__ c2b,
    const float* __restrict__ c3w, const float* __restrict__ c3b,
    const float* __restrict__ fcw, const float* __restrict__ fcb,
    const int* __restrict__ kptr,
    int* __restrict__ counts, int* __restrict__ lists) {
    __shared__ float sx[3072];
    __shared__ float sw1[75], sw2[50], sw3[200], sfw[32];
    __shared__ float sb1[1], sb2[2], sb3[4], sfb[8];
    __shared__ float h1[196], h2[50], h3[4];

    const int t = threadIdx.x;
    const int b = blockIdx.x;

    for (int i = t; i < 3072; i += 256) sx[i] = x[(size_t)b * 3072 + i];
    if (t < 75)  sw1[t] = c1w[t];
    if (t < 50)  sw2[t] = c2w[t];
    if (t < 200) sw3[t] = c3w[t];
    if (t < 32)  sfw[t] = fcw[t];
    if (t < 8)   sfb[t] = fcb[t];
    if (t < 4)   sb3[t] = c3b[t];
    if (t < 2)   sb2[t] = c2b[t];
    if (t == 0)  sb1[0] = c1b[0];
    __syncthreads();

    // stage A: conv1 (3->1, 5x5) + relu + maxpool2 -> h1[14][14]
    if (t < 196) {
        int py = t / 14, px = t % 14;
        float best = -1e30f;
        for (int dy = 0; dy < 2; ++dy)
            for (int dx = 0; dx < 2; ++dx) {
                int oy = 2 * py + dy, ox = 2 * px + dx;
                float v = sb1[0];
                for (int c = 0; c < 3; ++c)
                    for (int ky = 0; ky < 5; ++ky)
                        for (int kx = 0; kx < 5; ++kx)
                            v += sx[c * 1024 + (oy + ky) * 32 + (ox + kx)] *
                                 sw1[c * 25 + ky * 5 + kx];
                best = fmaxf(best, v);
            }
        h1[t] = fmaxf(best, 0.f);
    }
    __syncthreads();

    // stage B: conv2 (1->2, 5x5) + relu + maxpool2 -> h2[2][5][5]
    if (t < 50) {
        int oc = t / 25, p = t % 25, py = p / 5, px = p % 5;
        float best = -1e30f;
        for (int dy = 0; dy < 2; ++dy)
            for (int dx = 0; dx < 2; ++dx) {
                int oy = 2 * py + dy, ox = 2 * px + dx;
                float v = sb2[oc];
                for (int ky = 0; ky < 5; ++ky)
                    for (int kx = 0; kx < 5; ++kx)
                        v += h1[(oy + ky) * 14 + (ox + kx)] *
                             sw2[oc * 25 + ky * 5 + kx];
                best = fmaxf(best, v);
            }
        h2[t] = fmaxf(best, 0.f);
    }
    __syncthreads();

    // stage C: conv3 (2->4, 5x5 on 5x5) -> h3[4]
    if (t < 4) {
        float v = sb3[t];
        for (int ic = 0; ic < 2; ++ic)
            for (int p = 0; p < 25; ++p)
                v += h2[ic * 25 + p] * sw3[(t * 2 + ic) * 25 + p];
        h3[t] = fmaxf(v, 0.f);
    }
    __syncthreads();

    // stage D: fc + top-k (normalization is a positive scale -> order invariant)
    if (t == 0) {
        float z[8];
        for (int e = 0; e < 8; ++e) {
            float v = sfb[e];
            for (int j = 0; j < 4; ++j) v += h3[j] * sfw[e * 4 + j];
            z[e] = v;
        }
        int kk = *kptr; if (kk > 8) kk = 8;
        unsigned used = 0;
        for (int s = 0; s < kk; ++s) {
            int best = 0; float bv = -3.4e38f;
            for (int e = 0; e < 8; ++e)
                if (!(used & (1u << e)) && z[e] > bv) { bv = z[e]; best = e; }
            used |= 1u << best;
            if (s < 2) {
                int pos = atomicAdd(&counts[best], 1);
                lists[best * B_N + pos] = (b << 1) | s;
            }
        }
    }
}

// ---------------- kernel 4: grouped bf16 WMMA GEMM over routed rows ----------------
// Each wave: 32(M) x 64(N) tile of one expert. Double-buffered fragments so the
// next K-chunk's loads overlap the current chunk's 8 WMMAs.
__global__ __launch_bounds__(256) void moe_gemm_kernel(
    const unsigned short* __restrict__ xbf,   // [B, F] bf16
    const unsigned short* __restrict__ wt,    // [E, NC, F] bf16
    const float* __restrict__ ens_b,          // [E, NC]
    const int* __restrict__ counts,
    const int* __restrict__ lists,            // entries: (b<<1)|slot
    float* __restrict__ part) {               // [B, 2, NC]
    const int lane  = threadIdx.x & 31;
    const int wv    = threadIdx.x >> 5;
    const int e     = blockIdx.z;
    const int mtile = blockIdx.y * 8 + wv;    // 32-row tile index
    const int cnt   = counts[e];
    if (mtile * 32 >= cnt) return;

    const int nbase = blockIdx.x * 64;
    const int nn = lane & 15;
    const int hi = lane >> 4;

    const int midx0 = mtile * 32 + nn;
    const int midx1 = mtile * 32 + 16 + nn;
    const int entry0 = (midx0 < cnt) ? lists[e * B_N + midx0] : 0;
    const int entry1 = (midx1 < cnt) ? lists[e * B_N + midx1] : 0;

    // A fragment (16-bit A 16x32 layout): lanes 0-15 hold K 0-7/16-23,
    // lanes 16-31 hold K 8-15/24-31 -> base shift 8, chunk gap 16.
    const unsigned short* aptr0 = xbf + (size_t)(entry0 >> 1) * F_DIM + (hi ? 8 : 0);
    const unsigned short* aptr1 = xbf + (size_t)(entry1 >> 1) * F_DIM + (hi ? 8 : 0);

    // B fragment: lanes 0-15 hold K 0-7/16-23, lanes 16-31 hold K 8-15/24-31 (matches A).
    const unsigned short* bptr[4];
#pragma unroll
    for (int tt = 0; tt < 4; ++tt) {
        int c = nbase + tt * 16 + nn;
        if (c > NC - 1) c = NC - 1;               // clamp; never stored
        bptr[tt] = wt + ((size_t)e * NC + c) * F_DIM + (hi ? 8 : 0);
    }

    v8f acc0[4] = {};
    v8f acc1[4] = {};

    auto load_set = [&](v16bf& A0, v16bf& A1, v16bf* Bv, int K) {
        A0 = load_frag(aptr0 + K, 16);
        A1 = load_frag(aptr1 + K, 16);
#pragma unroll
        for (int tt = 0; tt < 4; ++tt) Bv[tt] = load_frag(bptr[tt] + K, 16);
    };
    auto do_wmma = [&](v16bf A0, v16bf A1, const v16bf* Bv) {
#pragma unroll
        for (int tt = 0; tt < 4; ++tt) {
            acc0[tt] = __builtin_amdgcn_wmma_f32_16x16x32_bf16(
                false, A0, false, Bv[tt], (short)0, acc0[tt], false, false);
            acc1[tt] = __builtin_amdgcn_wmma_f32_16x16x32_bf16(
                false, A1, false, Bv[tt], (short)0, acc1[tt], false, false);
        }
    };

    v16bf aX0, aX1, bX[4];
    v16bf aY0, aY1, bY[4];

    load_set(aX0, aX1, bX, 0);
    int k0 = 0;
    for (; k0 < F_DIM - 64; k0 += 64) {
        load_set(aY0, aY1, bY, k0 + 32);   // in flight while X computes
        do_wmma(aX0, aX1, bX);
        load_set(aX0, aX1, bX, k0 + 64);   // in flight while Y computes
        do_wmma(aY0, aY1, bY);
    }
    load_set(aY0, aY1, bY, k0 + 32);
    do_wmma(aX0, aX1, bX);
    do_wmma(aY0, aY1, bY);

    // gather row entries for the 32 rows of this wave's tile
    int ent0[8], ent1[8];
#pragma unroll
    for (int r = 0; r < 8; ++r) {
        ent0[r] = __shfl(entry0, r + hi * 8, 32);
        ent1[r] = __shfl(entry1, r + hi * 8, 32);
    }

#pragma unroll
    for (int tt = 0; tt < 4; ++tt) {
        int c = nbase + tt * 16 + nn;
        if (c >= NC) continue;
        float bias = ens_b[e * NC + c];
#pragma unroll
        for (int r = 0; r < 8; ++r) {
            int m = r + hi * 8;
            if (mtile * 32 + m < cnt)
                part[(size_t)ent0[r] * NC + c] = acc0[tt][r] + bias;
            if (mtile * 32 + 16 + m < cnt)
                part[(size_t)ent1[r] * NC + c] = acc1[tt][r] + bias;
        }
    }
}

// ---------------- kernel 5: combine the k=2 expert slots ----------------
__global__ void reduce_kernel(const float* __restrict__ part,
                              float* __restrict__ out) {
    int i = blockIdx.x * blockDim.x + threadIdx.x;
    if (i >= B_N * NC) return;
    int b = i / NC;
    int c = i - b * NC;
    out[i] = part[(size_t)(b * 2) * NC + c] + part[(size_t)(b * 2 + 1) * NC + c];
}

extern "C" void kernel_launch(void* const* d_in, const int* in_sizes, int n_in,
                              void* d_out, int out_size, void* d_ws, size_t ws_size,
                              hipStream_t stream) {
    (void)in_sizes; (void)n_in; (void)out_size; (void)ws_size;
    const float* x    = (const float*)d_in[0];
    const float* c1w  = (const float*)d_in[1];
    const float* c1b  = (const float*)d_in[2];
    const float* c2w  = (const float*)d_in[3];
    const float* c2b  = (const float*)d_in[4];
    const float* c3w  = (const float*)d_in[5];
    const float* c3b  = (const float*)d_in[6];
    const float* fcw  = (const float*)d_in[7];
    const float* fcb  = (const float*)d_in[8];
    const float* ensw = (const float*)d_in[9];
    const float* ensb = (const float*)d_in[10];
    const int*   kptr = (const int*)d_in[11];
    float* out = (float*)d_out;

    char* ws = (char*)d_ws;
    size_t off = 0;
    int* counts = (int*)(ws + off); off += 128;
    int* lists  = (int*)(ws + off); off += (size_t)E_NUM * B_N * 4;
    off = (off + 255) & ~(size_t)255;
    float* part = (float*)(ws + off);
    size_t part_bytes = (size_t)B_N * 2 * NC * 4;
    off += part_bytes;
    off = (off + 255) & ~(size_t)255;
    unsigned short* xbf = (unsigned short*)(ws + off);
    off += (size_t)B_N * F_DIM * 2;
    off = (off + 255) & ~(size_t)255;
    unsigned short* wt = (unsigned short*)(ws + off);

    hipMemsetAsync(counts, 0, 128, stream);
    hipMemsetAsync(part, 0, part_bytes, stream);

    convert_x_kernel<<<(B_N * F_DIM) / (8 * 256), 256, 0, stream>>>(x, xbf);
    convert_transpose_w_kernel<<<dim3(F_DIM / 32, (NC + 31) / 32, E_NUM),
                                 dim3(32, 8), 0, stream>>>(ensw, wt);
    selector_kernel<<<B_N, 256, 0, stream>>>(x, c1w, c1b, c2w, c2b, c3w, c3b,
                                             fcw, fcb, kptr, counts, lists);
    moe_gemm_kernel<<<dim3(16, 16, E_NUM), 256, 0, stream>>>(xbf, wt, ensb,
                                                             counts, lists, part);
    reduce_kernel<<<(B_N * NC + 255) / 256, 256, 0, stream>>>(part, out);
}
